// SelfAttention_46918222742031
// MI455X (gfx1250) — matrix operand order repaired and checked by reading, not compiled, not run on hardware
//
#include <hip/hip_runtime.h>
#include <hip/hip_bf16.h>

// ---------------------------------------------------------------------------
// Causal self-attention forward for MI455X (gfx1250), wave32 + WMMA bf16.
//   B=2, T=2048, E=2048, H=16, d=128.
// Pipeline: fp32->bf16 cast (+weight transpose, V repack), QKV GEMM (WMMA),
//           flash-attention (WMMA), output GEMM + bias (WMMA, fp32 out).
// Tile staging: double-buffered global_load_async_to_lds_b128 (ASYNCcnt),
// overlapping the async DMA of tile t+1 with WMMA on tile t.
// ---------------------------------------------------------------------------

#define USE_ASYNC_LDS 1

typedef __attribute__((ext_vector_type(16))) __bf16 v16bf;
typedef __attribute__((ext_vector_type(8)))  float  v8f;

constexpr int kDim   = 2048;
constexpr int kHeads = 16;
constexpr int kBatch = 2;
constexpr int kSeq   = 2048;
constexpr int kHd    = 128;                 // head dim
constexpr int kRows  = kBatch * kSeq;       // 4096
constexpr int kLdQKV = 3 * kDim;            // 6144

union AB { v16bf v; unsigned short s[16]; };

__device__ __forceinline__ unsigned short f2bf(float f) {
  union { float f; unsigned u; } x; x.f = f;
  unsigned u = x.u;
  return (unsigned short)((u + 0x7FFFu + ((u >> 16) & 1u)) >> 16);  // RNE
}

__device__ __forceinline__ v8f zero8() {
  v8f z;
#pragma unroll
  for (int i = 0; i < 8; ++i) z[i] = 0.0f;
  return z;
}

// --- async memory->LDS staging helpers (CDNA5 GLOBAL_LOAD_ASYNC_TO_LDS) ----
__device__ __forceinline__ unsigned lds_off32(const void* p) {
  // Generic pointers to LDS carry the byte offset in the low 32 bits.
  return (unsigned)(unsigned long long)p;
}

__device__ __forceinline__ void stage16(unsigned short* ldst,
                                        const unsigned short* gsrc) {
  // Copy 16 bf16 (32 bytes) global -> LDS; both 16B aligned.
#if USE_ASYNC_LDS
  unsigned l = lds_off32(ldst);
  unsigned long long g = (unsigned long long)gsrc;
  asm volatile("global_load_async_to_lds_b128 %0, %1, off"
               :: "v"(l), "v"(g) : "memory");
  asm volatile("global_load_async_to_lds_b128 %0, %1, off offset:16"
               :: "v"(l), "v"(g) : "memory");
#else
  ((uint4*)ldst)[0] = ((const uint4*)gsrc)[0];
  ((uint4*)ldst)[1] = ((const uint4*)gsrc)[1];
#endif
}

// Wait until at most `keep4` (0 or 4) async ops remain outstanding.  Async
// loads complete in order, so keeping the newest 4 means the previous tile
// (the one about to be consumed) has fully landed in LDS.
__device__ __forceinline__ void stage_wait(bool keep4) {
#if USE_ASYNC_LDS
#if defined(__has_builtin) && __has_builtin(__builtin_amdgcn_s_wait_asynccnt)
  if (keep4) __builtin_amdgcn_s_wait_asynccnt(4);
  else       __builtin_amdgcn_s_wait_asynccnt(0);
#else
  if (keep4) asm volatile("s_wait_asynccnt 0x4" ::: "memory");
  else       asm volatile("s_wait_asynccnt 0x0" ::: "memory");
#endif
#endif
}

__device__ __forceinline__ void wave_fence() {
#if defined(__has_builtin) && __has_builtin(__builtin_amdgcn_wave_barrier)
  __builtin_amdgcn_wave_barrier();   // intra-wave scheduling fence (free)
#else
  __syncthreads();
#endif
}

// ---------------------------------------------------------------------------
// fp32 -> bf16 cast (grid-stride)
// ---------------------------------------------------------------------------
__global__ void cvt_bf16(const float* __restrict__ in,
                         unsigned short* __restrict__ out, int n) {
  int i = blockIdx.x * blockDim.x + threadIdx.x;
  int stride = gridDim.x * blockDim.x;
  for (; i < n; i += stride) out[i] = f2bf(in[i]);
}

// ---------------------------------------------------------------------------
// fp32 [R][C] -> bf16 transposed [C][R]   (tiled through LDS)
// block (32,8), grid (C/32, R/32)
// ---------------------------------------------------------------------------
__global__ __launch_bounds__(256) void cvt_transpose(
    const float* __restrict__ in, unsigned short* __restrict__ outT,
    int R, int C) {
  __shared__ unsigned short tile[32][33];
  const int c0 = blockIdx.x * 32, r0 = blockIdx.y * 32;
  const int tx = threadIdx.x, ty = threadIdx.y;
#pragma unroll
  for (int i = ty; i < 32; i += 8)
    tile[i][tx] = f2bf(in[(long)(r0 + i) * C + c0 + tx]);
  __syncthreads();
#pragma unroll
  for (int i = ty; i < 32; i += 8)
    outT[(long)(c0 + i) * R + r0 + tx] = tile[tx][i];
}

// ---------------------------------------------------------------------------
// Repack V (bf16 slice of qkv, [b,t,h,d]) -> vT [b,h,d,T]
// block (32,8), grid (T/32, d/32, B*H)
// ---------------------------------------------------------------------------
__global__ __launch_bounds__(256) void repack_vT(
    const unsigned short* __restrict__ qkv, unsigned short* __restrict__ vT) {
  __shared__ unsigned short tile[32][33];
  const int bh = blockIdx.z, b = bh >> 4, h = bh & 15;
  const int t0 = blockIdx.x * 32, d0 = blockIdx.y * 32;
  const int tx = threadIdx.x, ty = threadIdx.y;
  const unsigned short* Vp = qkv + (long)b * kSeq * kLdQKV + 2 * kDim + h * kHd;
#pragma unroll
  for (int i = ty; i < 32; i += 8)
    tile[i][tx] = Vp[(long)(t0 + i) * kLdQKV + d0 + tx];   // tile[t][d]
  __syncthreads();
#pragma unroll
  for (int i = ty; i < 32; i += 8)
    vT[((long)bh * kHd + d0 + i) * kSeq + t0 + tx] = tile[tx][i];
}

// ---------------------------------------------------------------------------
// bf16 GEMM: C[M,N] = A[M,K] * Bt[N,K]^T  (B pre-transposed).
// Block tile 128x128, BK=32, 8 waves, wave tile 64x32 (4x2 WMMA accums).
// Double-buffered async LDS staging.  OUT_BF16: bf16 store, else fp32+bias.
// ---------------------------------------------------------------------------
template <bool OUT_BF16>
__global__ __launch_bounds__(256) void gemm_bf16(
    const unsigned short* __restrict__ A, const unsigned short* __restrict__ Bt,
    void* __restrict__ Cout, const float* __restrict__ bias,
    int M, int N, int K, int lda, int ldbT, int ldc) {
  __shared__ unsigned short sA[2][128 * 32];   // [m][k]
  __shared__ unsigned short sBt[2][128 * 32];  // [n][k]

  const int tid  = threadIdx.x;
  const int lane = tid & 31;
  const int wid  = tid >> 5;
  const int wm   = wid >> 2;      // 0..1 -> 64-row group
  const int wn   = wid & 3;       // 0..3 -> 32-col group
  const int lh   = lane & 15;
  const int hh   = lane >> 4;
  const int m0   = blockIdx.y * 128;
  const int n0   = blockIdx.x * 128;

  // Staging assignment: each thread copies one 16-element (32B) run per tile.
  const int srow = tid >> 1;            // 0..127
  const int scol = (tid & 1) * 16;      // 0 or 16

  auto stage = [&](int buf, int kk) {
    stage16(&sA[buf][srow * 32 + scol],
            &A[(long)(m0 + srow) * lda + kk + scol]);
    stage16(&sBt[buf][srow * 32 + scol],
            &Bt[(long)(n0 + srow) * ldbT + kk + scol]);
  };

  v8f acc[4][2];
#pragma unroll
  for (int mi = 0; mi < 4; ++mi)
#pragma unroll
    for (int ni = 0; ni < 2; ++ni) acc[mi][ni] = zero8();

  const int KT = K >> 5;
  stage(0, 0);
  for (int t = 0; t < KT; ++t) {
    const int cur = t & 1;
    const bool more = (t + 1 < KT);
    if (more) {
      stage(cur ^ 1, (t + 1) * 32);        // overlap next tile's DMA with math
      if (t + 2 < KT)                      // warm L2 two tiles ahead
        __builtin_prefetch(&A[(long)(m0 + srow) * lda + (t + 2) * 32 + scol],
                           0, 0);
    }
    stage_wait(more);
    __syncthreads();                       // tile t ready in all waves

    // B fragments (reused across mi): lane = column, half-wave = K group.
    AB bf[2];
#pragma unroll
    for (int ni = 0; ni < 2; ++ni) {
      int n = wn * 32 + ni * 16 + lh;
#pragma unroll
      for (int j = 0; j < 16; ++j) bf[ni].s[j] = sBt[cur][n * 32 + hh * 8 + (j & 7) + (j >> 3) * 16];
    }
#pragma unroll
    for (int mi = 0; mi < 4; ++mi) {
      AB af;  // A 16x32: lanes 0-15 K{0..7,16..23}, lanes 16-31 K{8..15,24..31}
      int r = wm * 64 + mi * 16 + lh;
#pragma unroll
      for (int j = 0; j < 8; ++j) {
        af.s[j]     = sA[cur][r * 32 + hh * 8 + j];
        af.s[8 + j] = sA[cur][r * 32 + hh * 8 + 16 + j];
      }
#pragma unroll
      for (int ni = 0; ni < 2; ++ni)
        acc[mi][ni] = __builtin_amdgcn_wmma_f32_16x16x32_bf16(
            false, af.v, false, bf[ni].v, (short)0, acc[mi][ni], false, false);
    }
    __syncthreads();   // all waves done reading buf `cur` before it is re-staged
  }

  // Epilogue.  C layout: m = r + 8*hh, n = lh (per 16x16 tile).
#pragma unroll
  for (int mi = 0; mi < 4; ++mi)
#pragma unroll
    for (int ni = 0; ni < 2; ++ni)
#pragma unroll
      for (int r = 0; r < 8; ++r) {
        long m = m0 + wm * 64 + mi * 16 + r + 8 * hh;
        long n = n0 + wn * 32 + ni * 16 + lh;
        float v = acc[mi][ni][r];
        if constexpr (OUT_BF16) {
          ((unsigned short*)Cout)[m * ldc + n] = f2bf(v);
        } else {
          ((float*)Cout)[m * ldc + n] = v + bias[n];
        }
      }
}

// ---------------------------------------------------------------------------
// Flash attention forward (causal).  Grid: (T/128, B*H), 256 threads.
// Each wave owns 16 query rows; block stages 32-key K/V chunks in LDS with
// double-buffered async copies.  Softmax denominator via P @ ones WMMA.
// qkv: bf16 [B*T, 3E] (Q | K | V);  vT: bf16 [B,H,d,T];  outb: bf16 [B*T, E].
// ---------------------------------------------------------------------------
__global__ __launch_bounds__(256) void attn_fwd(
    const unsigned short* __restrict__ qkv,
    const unsigned short* __restrict__ vT,
    unsigned short* __restrict__ outb) {
  __shared__ unsigned short sK[2][32 * 128];   // [key][d]
  __shared__ unsigned short sVt[2][128 * 32];  // [d][key]
  __shared__ unsigned short sP[8][16 * 32];    // per-wave P scratch [m][k]

  const int tid  = threadIdx.x;
  const int lane = tid & 31;
  const int wid  = tid >> 5;
  const int lh   = lane & 15;
  const int hh   = lane >> 4;
  const int qb   = blockIdx.x;
  const int bh   = blockIdx.y;
  const int b    = bh >> 4;
  const int h    = bh & 15;

  const long rowbase = (long)b * kSeq;
  const unsigned short* Qp  = qkv + rowbase * kLdQKV + h * kHd;
  const unsigned short* Kp  = Qp + kDim;
  const unsigned short* Vtp = vT + (long)bh * kHd * kSeq;   // [d][T]

  const int   qrow0 = qb * 128 + wid * 16;
  const int   diag0 = qb * 128;               // first k0 needing a mask
  const float SL2E  = 0.08838834764831845f * 1.4426950408889634f; // scale*log2e

  // Staging assignment (one 32B run per thread per buffer).
  const int krow = tid >> 3, kcol = (tid & 7) * 16;  // sK: [32][128]
  const int vrow = tid >> 1, vcol = (tid & 1) * 16;  // sVt: [128][32]

  auto stage = [&](int buf, int k0s) {
    stage16(&sK[buf][krow * 128 + kcol],
            &Kp[(long)(k0s + krow) * kLdQKV + kcol]);
    stage16(&sVt[buf][vrow * 32 + vcol],
            &Vtp[(long)vrow * kSeq + k0s + vcol]);
  };

  // Q fragments: 16 rows x 128 d as 4 A-fragments (16x32 each).
  AB aq[4];
  {
    long r = qrow0 + lh;
#pragma unroll
    for (int c = 0; c < 4; ++c) {
      int kb = c * 32 + hh * 8;
#pragma unroll
      for (int j = 0; j < 8; ++j) {
        aq[c].s[j]     = Qp[r * kLdQKV + kb + j];
        aq[c].s[8 + j] = Qp[r * kLdQKV + kb + 16 + j];
      }
    }
  }

  // Constant all-ones B fragment: O9 = P @ 1 accumulates the softmax
  // denominator (every column equals the row sum).
  AB onesf;
#pragma unroll
  for (int j = 0; j < 16; ++j) onesf.s[j] = 0x3F80;  // bf16 1.0

  v8f o[8], o9;
#pragma unroll
  for (int i = 0; i < 8; ++i) o[i] = zero8();
  o9 = zero8();
  float mst[8];
#pragma unroll
  for (int r = 0; r < 8; ++r) mst[r] = -1e30f;

  const int KT = (qb * 128 + 128) >> 5;
  stage(0, 0);
  for (int t = 0; t < KT; ++t) {
    const int  cur  = t & 1;
    const int  k0   = t * 32;
    const bool more = (t + 1 < KT);
    if (more) stage(cur ^ 1, (t + 1) * 32);
    stage_wait(more);
    __syncthreads();                       // chunk t ready

    // S = Q * K^T for two 16-key tiles (GEMM K-dim = head dim, 4x32).
    float p0[8], p1[8];
#pragma unroll
    for (int tt = 0; tt < 2; ++tt) {
      v8f st = zero8();
#pragma unroll
      for (int c = 0; c < 4; ++c) {
        AB bk;  // B 32x16: lane = key column, d runs {hh*8..+7, 16+hh*8..+7}
        int key  = tt * 16 + lh;
        int doff = c * 32 + hh * 8;
#pragma unroll
        for (int j = 0; j < 16; ++j) bk.s[j] = sK[cur][key * 128 + doff + (j & 7) + (j >> 3) * 16];
        st = __builtin_amdgcn_wmma_f32_16x16x32_bf16(
            false, aq[c].v, false, bk.v, (short)0, st, false, false);
      }
#pragma unroll
      for (int r = 0; r < 8; ++r) { if (tt) p1[r] = st[r]; else p0[r] = st[r]; }
    }

    if (k0 >= diag0) {                     // diagonal block: causal mask
#pragma unroll
      for (int r = 0; r < 8; ++r) {
        int irow = qrow0 + r + 8 * hh;
        if (k0 + lh      > irow) p0[r] = -1e30f;
        if (k0 + 16 + lh > irow) p1[r] = -1e30f;
      }
    }

    // Online softmax update (raw-score units; 1/sqrt(d) folded into SL2E).
#pragma unroll
    for (int r = 0; r < 8; ++r) {
      float mx = fmaxf(p0[r], p1[r]);
      for (int off = 1; off < 16; off <<= 1)
        mx = fmaxf(mx, __shfl_xor(mx, off, 32));
      float mnew = fmaxf(mst[r], mx);
      float corr = exp2f((mst[r] - mnew) * SL2E);
      mst[r] = mnew;
      p0[r] = exp2f((p0[r] - mnew) * SL2E);
      p1[r] = exp2f((p1[r] - mnew) * SL2E);
#pragma unroll
      for (int ni = 0; ni < 8; ++ni) o[ni][r] *= corr;
      o9[r] *= corr;
    }

    // C-layout -> A-layout relayout of P through per-wave LDS scratch.
    // DS ops are in-order within a wave: a scheduling fence suffices.
#pragma unroll
    for (int r = 0; r < 8; ++r) {
      int m = r + 8 * hh;
      sP[wid][m * 32 + lh]      = f2bf(p0[r]);
      sP[wid][m * 32 + 16 + lh] = f2bf(p1[r]);
    }
    wave_fence();

    AB ap;
#pragma unroll
    for (int j = 0; j < 8; ++j) {
      ap.s[j]     = sP[wid][lh * 32 + hh * 8 + j];
      ap.s[8 + j] = sP[wid][lh * 32 + hh * 8 + 16 + j];
    }
    // O += P * V  (K-dim = 32 keys, 8 column tiles over d=128) + denominator.
#pragma unroll
    for (int ni = 0; ni < 8; ++ni) {
      AB bv;
      int dcol = ni * 16 + lh;
#pragma unroll
      for (int j = 0; j < 16; ++j) bv.s[j] = sVt[cur][dcol * 32 + hh * 8 + (j & 7) + (j >> 3) * 16];
      o[ni] = __builtin_amdgcn_wmma_f32_16x16x32_bf16(
          false, ap.v, false, bv.v, (short)0, o[ni], false, false);
    }
    o9 = __builtin_amdgcn_wmma_f32_16x16x32_bf16(
        false, ap.v, false, onesf.v, (short)0, o9, false, false);

    __syncthreads();   // all waves done reading buf `cur` before re-staging
  }

  // Normalize by the WMMA-accumulated denominator and store bf16 [B*T, E].
#pragma unroll
  for (int r = 0; r < 8; ++r) {
    float inv = 1.0f / o9[r];
    long  m   = qrow0 + r + 8 * hh;
#pragma unroll
    for (int ni = 0; ni < 8; ++ni)
      outb[(rowbase + m) * kDim + h * kHd + ni * 16 + lh] =
          f2bf(o[ni][r] * inv);
  }
}

// ---------------------------------------------------------------------------
// Host launcher
// ---------------------------------------------------------------------------
extern "C" void kernel_launch(void* const* d_in, const int* in_sizes, int n_in,
                              void* d_out, int out_size, void* d_ws, size_t ws_size,
                              hipStream_t stream) {
  const float* x     = (const float*)d_in[0];
  const float* w_qkv = (const float*)d_in[1];
  const float* w_out = (const float*)d_in[2];
  const float* b_out = (const float*)d_in[3];
  float* out = (float*)d_out;

  char* ws = (char*)d_ws;
  size_t off = 0;
  auto carve = [&](size_t bytes) -> void* {
    void* p = ws + off;
    off = (off + bytes + 255) & ~(size_t)255;
    return p;
  };
  unsigned short* xb     = (unsigned short*)carve((size_t)kRows * kDim * 2);
  unsigned short* wqkvT  = (unsigned short*)carve((size_t)kDim * 3 * kDim * 2);
  unsigned short* woutT  = (unsigned short*)carve((size_t)kDim * kDim * 2);
  unsigned short* qkvb   = (unsigned short*)carve((size_t)kRows * 3 * kDim * 2);
  unsigned short* vTb    = (unsigned short*)carve((size_t)kRows * kDim * 2);
  unsigned short* attnb  = (unsigned short*)carve((size_t)kRows * kDim * 2);

  // Casts / transposes (bandwidth-trivial, one pass each).
  cvt_bf16<<<2048, 256, 0, stream>>>(x, xb, kRows * kDim);
  cvt_transpose<<<dim3(3 * kDim / 32, kDim / 32), dim3(32, 8), 0, stream>>>(
      w_qkv, wqkvT, kDim, 3 * kDim);
  cvt_transpose<<<dim3(kDim / 32, kDim / 32), dim3(32, 8), 0, stream>>>(
      w_out, woutT, kDim, kDim);

  // qkv = x @ w_qkv : bf16 [4096, 6144]
  gemm_bf16<true><<<dim3(3 * kDim / 128, kRows / 128), 256, 0, stream>>>(
      xb, wqkvT, qkvb, nullptr, kRows, 3 * kDim, kDim, kDim, kDim, 3 * kDim);

  // V slice -> [b,h,d,T] for contiguous V^T staging in attention.
  repack_vT<<<dim3(kSeq / 32, kHd / 32, kBatch * kHeads), dim3(32, 8), 0,
              stream>>>(qkvb, vTb);

  // flash attention -> bf16 [4096, 2048]
  attn_fwd<<<dim3(kSeq / 128, kBatch * kHeads), 256, 0, stream>>>(qkvb, vTb,
                                                                  attnb);

  // out = attn @ w_out + b_out : fp32 [4096, 2048]
  gemm_bf16<false><<<dim3(kDim / 128, kRows / 128), 256, 0, stream>>>(
      attnb, woutT, out, b_out, kRows, kDim, kDim, kDim, kDim, kDim);
}
